// OscSpecSinkhornLoss_86852828660258
// MI455X (gfx1250) — compile-verified, matrix operand order checked by reading, not run on hardware
//
#include <hip/hip_runtime.h>

typedef __attribute__((ext_vector_type(16))) _Float16 v16h;
typedef __attribute__((ext_vector_type(8)))  float    v8f;

#define NA      64
#define NB      513
#define NBPAD   544          // 17 WMMA k-tiles of 32
#define CSTRIDE 548          // row stride: 548 mod 64 = 36 -> 16 rows spread across all banks
#define EPS     0.01f
#define INV_EPS 100.0f
#define NITER   5

// dynamic-LDS layout (float offsets)
#define OFF_CHAT 0
#define OFF_VHAT (NA * CSTRIDE)          // 35072
#define OFF_NU   (OFF_VHAT + NBPAD)
#define OFF_BINF (OFF_NU + NB)
#define OFF_UHAT (OFF_BINF + NBPAD)
#define OFF_MU   (OFF_UHAT + NA)
#define OFF_FR   (OFF_MU + NA)
#define OFF_RED  (OFF_FR + NA)
#define OFF_UMAX (OFF_RED + 256)         // [2][4][16]
#define OFF_USUM (OFF_UMAX + 128)        // [2][4][16]
#define SMEM_FLOATS (OFF_USUM + 128)
#define SMEM_BYTES  (SMEM_FLOATS * 4)    // ~149.5 KB of the 320 KB/WGP

__global__ __launch_bounds__(256) void sinkhorn_frame_kernel(
    const float* __restrict__ freqs_g,   // (N,64)
    const float* __restrict__ amps_g,    // (N,64)
    const float* __restrict__ targ_g,    // (N,513)
    const float* __restrict__ binf_g,    // (513)
    float* __restrict__ frame_out)       // (N,)
{
  extern __shared__ float smem[];
  float* chat  = smem + OFF_CHAT;   // chat[a*CSTRIDE+b] = -|f_a - bin_b| / eps  (pad: -1e32)
  float* vhat  = smem + OFF_VHAT;   // v / eps, padded with 0
  float* nuW   = smem + OFF_NU;
  float* binf  = smem + OFF_BINF;
  float* uhat  = smem + OFF_UHAT;   // u / eps
  float* muW   = smem + OFF_MU;
  float* frW   = smem + OFF_FR;
  float* red   = smem + OFF_RED;
  float* umaxS = smem + OFF_UMAX;
  float* usumS = smem + OFF_USUM;

  const int tid = threadIdx.x;
  const int n   = blockIdx.x;

  // ---------------- phase A: loads + mu/nu normalization --------------------
  for (int b = tid; b < NBPAD; b += 256) {
    binf[b] = (b < NB) ? binf_g[b] : 0.f;
    vhat[b] = 0.f;                                   // v = 0 initially; pads stay 0 forever
  }
  float part = 0.f;
  for (int b = tid; b < NB; b += 256) {
    float t = targ_g[n * NB + b];
    t = fminf(fmaxf(t, 0.f), 1e9f) + 1e-9f;
    nuW[b] = t; part += t;
  }
  red[tid] = part; __syncthreads();
  for (int s = 128; s > 0; s >>= 1) { if (tid < s) red[tid] += red[tid + s]; __syncthreads(); }
  const float lsumT = __logf(red[0]);
  __syncthreads();

  float pa = 0.f;
  if (tid < NA) {
    float a = amps_g[n * NA + tid];
    a = fminf(fmaxf(a, 0.f), 1e9f) + 1e-9f;
    muW[tid] = a; pa = a;
    frW[tid] = freqs_g[n * NA + tid];
  }
  red[tid] = pa; __syncthreads();
  for (int s = 128; s > 0; s >>= 1) { if (tid < s) red[tid] += red[tid + s]; __syncthreads(); }
  const float lsumA = __logf(red[0]);
  __syncthreads();

  for (int b = tid; b < NB; b += 256) nuW[b] = __logf(nuW[b]) - lsumT;
  if (tid < NA) muW[tid] = __logf(muW[tid]) - lsumA;
  __syncthreads();

  // ---------------- phase B: precompute scaled cost (iteration-invariant) ---
  for (int a = 0; a < NA; ++a) {
    const float f = frW[a];
    for (int b = tid; b < NBPAD; b += 256)
      chat[a * CSTRIDE + b] = (b < NB) ? (-fabsf(f - binf[b]) * INV_EPS) : -1e32f;
  }
  __syncthreads();

  // ---------------- wave layout for the WMMA u-update -----------------------
  const int lane = tid & 31;
  const int wave = tid >> 5;
  const int rt   = wave & 3;          // row tile: a in [16*rt, 16*rt+16)
  const int kg   = wave >> 2;         // b-range half
  const int half = lane >> 4;         // A-matrix lane-half (K sub-block)
  const int row  = lane & 15;
  const int rowg = rt * 16 + row;
  const int tiBeg = kg ? 9 : 0;
  const int tiEnd = kg ? 17 : 9;
  const float* crow = chat + rowg * CSTRIDE;

  v16h ones;
  #pragma unroll
  for (int j = 0; j < 16; ++j) ones[j] = (_Float16)1.0f;

  for (int it = 0; it < NITER; ++it) {
    // ===== u update: uhat_a = mu_a - lse_b(vhat_b + chat_ab) =====
    // pass 1: per-row max (branch-free b128 tile loads)
    float m = -3.0e38f;
    for (int ti = tiBeg; ti < tiEnd; ++ti) {
      const int b0 = ti * 32 + half * 8;
      float4 c0 = *(const float4*)(crow + b0);
      float4 c1 = *(const float4*)(crow + b0 + 4);
      float4 c2 = *(const float4*)(crow + b0 + 16);
      float4 c3 = *(const float4*)(crow + b0 + 20);
      float4 w0 = *(const float4*)(vhat + b0);
      float4 w1 = *(const float4*)(vhat + b0 + 4);
      float4 w2 = *(const float4*)(vhat + b0 + 16);
      float4 w3 = *(const float4*)(vhat + b0 + 20);
      m = fmaxf(m, fmaxf(fmaxf(c0.x + w0.x, c0.y + w0.y), fmaxf(c0.z + w0.z, c0.w + w0.w)));
      m = fmaxf(m, fmaxf(fmaxf(c1.x + w1.x, c1.y + w1.y), fmaxf(c1.z + w1.z, c1.w + w1.w)));
      m = fmaxf(m, fmaxf(fmaxf(c2.x + w2.x, c2.y + w2.y), fmaxf(c2.z + w2.z, c2.w + w2.w)));
      m = fmaxf(m, fmaxf(fmaxf(c3.x + w3.x, c3.y + w3.y), fmaxf(c3.z + w3.z, c3.w + w3.w)));
    }
    m = fmaxf(m, __shfl_xor(m, 16, 32));            // rows split across lanes L / L+16
    if (lane < 16) umaxS[kg * 64 + rt * 16 + row] = m;
    __syncthreads();
    m = fmaxf(umaxS[rt * 16 + row], umaxS[64 + rt * 16 + row]);

    // pass 2: exp + ones-matmul row-sum on the matrix pipe
    v8f acc = {};
    for (int ti = tiBeg; ti < tiEnd; ++ti) {
      const int b0 = ti * 32 + half * 8;
      float4 c0 = *(const float4*)(crow + b0);
      float4 c1 = *(const float4*)(crow + b0 + 4);
      float4 c2 = *(const float4*)(crow + b0 + 16);
      float4 c3 = *(const float4*)(crow + b0 + 20);
      float4 w0 = *(const float4*)(vhat + b0);
      float4 w1 = *(const float4*)(vhat + b0 + 4);
      float4 w2 = *(const float4*)(vhat + b0 + 16);
      float4 w3 = *(const float4*)(vhat + b0 + 20);
      v16h av;
      av[0]  = (_Float16)__expf(c0.x + w0.x - m);   // K = 8*half + j
      av[1]  = (_Float16)__expf(c0.y + w0.y - m);
      av[2]  = (_Float16)__expf(c0.z + w0.z - m);
      av[3]  = (_Float16)__expf(c0.w + w0.w - m);
      av[4]  = (_Float16)__expf(c1.x + w1.x - m);
      av[5]  = (_Float16)__expf(c1.y + w1.y - m);
      av[6]  = (_Float16)__expf(c1.z + w1.z - m);
      av[7]  = (_Float16)__expf(c1.w + w1.w - m);
      av[8]  = (_Float16)__expf(c2.x + w2.x - m);   // K = 16 + 8*half + j
      av[9]  = (_Float16)__expf(c2.y + w2.y - m);
      av[10] = (_Float16)__expf(c2.z + w2.z - m);
      av[11] = (_Float16)__expf(c2.w + w2.w - m);
      av[12] = (_Float16)__expf(c3.x + w3.x - m);
      av[13] = (_Float16)__expf(c3.y + w3.y - m);
      av[14] = (_Float16)__expf(c3.z + w3.z - m);
      av[15] = (_Float16)__expf(c3.w + w3.w - m);
      acc = __builtin_amdgcn_wmma_f32_16x16x32_f16(false, av, false, ones,
                                                   (short)0, acc, false, false);
    }
    if (lane == 0) {
      #pragma unroll
      for (int r = 0; r < 8; ++r) usumS[kg * 64 + rt * 16 + r] = acc[r];       // rows 0..7
    }
    if (lane == 16) {
      #pragma unroll
      for (int r = 0; r < 8; ++r) usumS[kg * 64 + rt * 16 + 8 + r] = acc[r];   // rows 8..15
    }
    __syncthreads();
    if (tid < NA) {
      const int rr = tid & 15, rtt = tid >> 4;
      const float mm = fmaxf(umaxS[rtt * 16 + rr], umaxS[64 + rtt * 16 + rr]);
      const float ss = usumS[rtt * 16 + rr] + usumS[64 + rtt * 16 + rr];
      uhat[tid] = muW[tid] - (mm + __logf(ss));
    }
    __syncthreads();

    // ===== v update: vhat_b = nu_b - lse_a(uhat_a + chat_ab) =====
    for (int b = tid; b < NB; b += 256) {
      const float* ccol = chat + b;
      float mv = -3.0e38f;
      #pragma unroll 4
      for (int a = 0; a < NA; ++a)
        mv = fmaxf(mv, uhat[a] + ccol[a * CSTRIDE]);
      float sum = 0.f;
      #pragma unroll 4
      for (int a = 0; a < NA; ++a)
        sum += __expf(uhat[a] + ccol[a * CSTRIDE] - mv);
      vhat[b] = nuW[b] - (mv + __logf(sum));
    }
    __syncthreads();
  }

  // ===== loss: sum exp(uhat+vhat+chat) * cost, cost = -eps*chat =====
  float accL = 0.f;
  for (int b = tid; b < NB; b += 256) {
    const float vb = vhat[b];
    const float* ccol = chat + b;
    #pragma unroll 4
    for (int a = 0; a < NA; ++a) {
      const float ch = ccol[a * CSTRIDE];
      accL += __expf(uhat[a] + vb + ch) * ch;
    }
  }
  red[tid] = accL; __syncthreads();
  for (int s = 128; s > 0; s >>= 1) { if (tid < s) red[tid] += red[tid + s]; __syncthreads(); }
  if (tid == 0) frame_out[n] = -EPS * red[0];
}

__global__ __launch_bounds__(256) void mean_kernel(const float* __restrict__ fo,
                                                   float* __restrict__ out, int n) {
  __shared__ float red[256];
  float a = 0.f;
  for (int i = threadIdx.x; i < n; i += 256) a += fo[i];
  red[threadIdx.x] = a; __syncthreads();
  for (int s = 128; s > 0; s >>= 1) {
    if (threadIdx.x < s) red[threadIdx.x] += red[threadIdx.x + s];
    __syncthreads();
  }
  if (threadIdx.x == 0) out[0] = red[0] / (float)n;
}

extern "C" void kernel_launch(void* const* d_in, const int* in_sizes, int n_in,
                              void* d_out, int out_size, void* d_ws, size_t ws_size,
                              hipStream_t stream) {
  const float* freqs = (const float*)d_in[0];   // (N,64) f32
  const float* amps  = (const float*)d_in[1];   // (N,64) f32
  const float* targ  = (const float*)d_in[2];   // (N,513) f32
  const float* binf  = (const float*)d_in[3];   // (513) f32
  const int N = in_sizes[0] / NA;               // 2048 frames

  // allow >64KB dynamic LDS (320KB/WGP on CDNA5); idempotent, capture-safe
  (void)hipFuncSetAttribute((const void*)sinkhorn_frame_kernel,
                            hipFuncAttributeMaxDynamicSharedMemorySize, SMEM_BYTES);

  float* frame_out = (float*)d_ws;              // N floats of scratch
  sinkhorn_frame_kernel<<<N, 256, SMEM_BYTES, stream>>>(freqs, amps, targ, binf, frame_out);
  mean_kernel<<<1, 256, 0, stream>>>(frame_out, (float*)d_out, N);
}
